// GaussianUpdate_34986803593232
// MI455X (gfx1250) — matrix operand order repaired and checked by reading, not compiled, not run on hardware
//
#include <hip/hip_runtime.h>

typedef __attribute__((ext_vector_type(16))) _Float16     v16h;
typedef __attribute__((ext_vector_type(8)))  float        v8f;
typedef __attribute__((ext_vector_type(4)))  unsigned int u32x4;
typedef __attribute__((ext_vector_type(4)))  int          i32x4;

typedef __attribute__((address_space(1))) i32x4 g_i32x4;   // global int4
typedef __attribute__((address_space(3))) i32x4 l_i32x4;   // LDS int4

// CDNA5 async global->LDS copy path (ASYNCcnt-tracked), guarded so the file
// compiles on toolchains that don't expose the builtins.
#if defined(__has_builtin)
#  if __has_builtin(__builtin_amdgcn_global_load_async_to_lds_b128) && \
      __has_builtin(__builtin_amdgcn_s_wait_asynccnt)
#    define USE_ASYNC_LDS 1
#  endif
#endif
#ifndef USE_ASYNC_LDS
#  define USE_ASYNC_LDS 0
#endif

__device__ __forceinline__ void async_copy_fence() {
#if USE_ASYNC_LDS
  __builtin_amdgcn_s_wait_asynccnt(0);
#endif
}

#define NCAM   6
#define LPIX   1680          // 28*60
#define DD     256
#define NPTS   8
#define MROWS  20160         // 2*6*1680
#define KAGG   2048          // 8*256
#define IMGW   60
#define IMGH   28

#define FEATS_N 5160960      // 2*10080*256
#define MEAN_N  60480        // 2*10080*3
#define COV_N   181440       // 2*10080*9

__device__ __forceinline__ unsigned pcg_hash(unsigned x) {
  x = x * 747796405u + 2891336453u;
  unsigned w = ((x >> ((x >> 28u) + 4u)) ^ x) * 277803737u;
  return (w >> 22u) ^ w;
}
__device__ __forceinline__ float u01(unsigned s) {
  return ((float)(pcg_hash(s) >> 8) + 0.5f) * (1.0f / 16777216.0f);
}
__device__ __forceinline__ int clampi(int v, int lo, int hi) {
  return v < lo ? lo : (v > hi ? hi : v);
}

// ---------------------------------------------------------------------------
// Weight conversion: fp32 row-major (K x Nsrc) -> f16 WMMA-fragment layout.
// Per (ktile, ntile) 16x16x32 B-tile, 512 halves stored so lane L's 16 halves
// are contiguous: dst[(kt*nTiles+nt)*512 + lane*16 + h],
// lane = (k%32)/16*16 + n%16, h = k%16.  Pads columns [Nsrc, Npad) with 0.
// ---------------------------------------------------------------------------
__global__ void convert_wfrag(const float* __restrict__ W, _Float16* __restrict__ dst,
                              int K, int Nsrc, int Npad) {
  int idx = blockIdx.x * blockDim.x + threadIdx.x;
  if (idx >= K * Npad) return;
  int k = idx / Npad, n = idx - k * Npad;
  float v = (n < Nsrc) ? W[(size_t)k * Nsrc + n] : 0.0f;
  int kt = k >> 5, krem = k & 31;
  int khalf = krem >> 4, h = krem & 15;
  int nt = n >> 4, ln = n & 15;
  int lane = khalf * 16 + ln;
  dst[((size_t)kt * (Npad >> 4) + nt) * 512 + (size_t)lane * 16 + h] = (_Float16)v;
}

// ---------------------------------------------------------------------------
// Init: transpose features (BN,C,H,W) -> row-major fp32 accumulator buffer;
// copy means3D / cov3D into working buffers.
// ---------------------------------------------------------------------------
__global__ __launch_bounds__(256) void init_kernel(
    const float* __restrict__ features, const float* __restrict__ means3D,
    const float* __restrict__ cov3D, float* __restrict__ feats,
    float* __restrict__ meanbuf, float* __restrict__ covbuf) {
  int row = blockIdx.x;
  int bn = row / LPIX, l = row - bn * LPIX;
  int c = threadIdx.x;
  feats[(size_t)row * DD + c] = features[((size_t)bn * DD + c) * LPIX + l];
  if (c < 3) meanbuf[(size_t)row * 3 + c] = means3D[(size_t)row * 3 + c];
  if (c < 9) covbuf[(size_t)row * 9 + c] = cov3D[(size_t)row * 9 + c];
}

// ---------------------------------------------------------------------------
// Gather: sample 8 MVN points (Cholesky + Box-Muller), project, bilinear
// sample img_features, write f16 staging matrix S (MROWS x 2048) + pts_cam.
// ---------------------------------------------------------------------------
__global__ __launch_bounds__(256) void gather_kernel(
    const float* __restrict__ img_features, const float* __restrict__ meanbuf,
    const float* __restrict__ covbuf, const float* __restrict__ lidar2img,
    _Float16* __restrict__ S, float* __restrict__ pts_out, int layer) {
  int row = blockIdx.x;
  int bn = row / LPIX;
  int t = threadIdx.x;
  int p = t >> 5;
  int lane = t & 31;

  const float* mu = meanbuf + (size_t)row * 3;
  const float* cv = covbuf + (size_t)row * 9;
  float a11 = cv[0], a21 = cv[3], a22 = cv[4], a31 = cv[6], a32 = cv[7], a33 = cv[8];
  float l11 = sqrtf(fmaxf(a11, 1e-12f));
  float inv11 = 1.0f / l11;
  float l21 = a21 * inv11, l31 = a31 * inv11;
  float l22 = sqrtf(fmaxf(a22 - l21 * l21, 1e-12f));
  float l32 = (a32 - l31 * l21) / l22;
  float l33 = sqrtf(fmaxf(a33 - l31 * l31 - l32 * l32, 1e-12f));

  unsigned base = 0x9E3779B9u * (unsigned)(layer + 1) ^
                  (0x85EBCA6Bu * (unsigned)row) ^ (0xC2B2AE35u * (unsigned)p);
  float u1 = u01(base + 11u), u2 = u01(base + 22u);
  float u3 = u01(base + 33u), u4 = u01(base + 44u);
  float r1 = sqrtf(fmaxf(-2.0f * __logf(u1), 0.0f));
  float r2 = sqrtf(fmaxf(-2.0f * __logf(u3), 0.0f));
  float e0 = r1 * __cosf(6.28318530718f * u2);
  float e1 = r1 * __sinf(6.28318530718f * u2);
  float e2 = r2 * __cosf(6.28318530718f * u4);

  float px = mu[0] + l11 * e0;
  float py = mu[1] + l21 * e0 + l22 * e1;
  float pz = mu[2] + l31 * e0 + l32 * e1 + l33 * e2;

  const float* Pm = lidar2img + (size_t)bn * 16;
  float cx = Pm[0] * px + Pm[1] * py + Pm[2] * pz + Pm[3];
  float cy = Pm[4] * px + Pm[5] * py + Pm[6] * pz + Pm[7];
  float cz = Pm[8] * px + Pm[9] * py + Pm[10] * pz + Pm[11];
  float homo = fmaxf(cz, 1e-6f);
  float uu = cx / homo, vv = cy / homo;
  if (lane == 0) {
    float* q = pts_out + (size_t)layer * MROWS * NPTS * 2 + (size_t)row * NPTS * 2 + p * 2;
    q[0] = uu; q[1] = vv;
  }
  float x = uu * (60.0f / 480.0f) - 0.5f;
  float y = vv * (28.0f / 224.0f) - 0.5f;
  float xf = floorf(x), yf = floorf(y);
  float wx = x - xf, wy = y - yf;
  int ix0 = (int)xf, iy0 = (int)yf;
  int ix1 = ix0 + 1, iy1 = iy0 + 1;
  float vx0 = (ix0 >= 0 && ix0 < IMGW) ? 1.f : 0.f;
  float vx1 = (ix1 >= 0 && ix1 < IMGW) ? 1.f : 0.f;
  float vy0 = (iy0 >= 0 && iy0 < IMGH) ? 1.f : 0.f;
  float vy1 = (iy1 >= 0 && iy1 < IMGH) ? 1.f : 0.f;
  float w00 = (1.f - wx) * (1.f - wy) * vx0 * vy0;
  float w10 = wx * (1.f - wy) * vx1 * vy0;
  float w01 = (1.f - wx) * wy * vx0 * vy1;
  float w11 = wx * wy * vx1 * vy1;
  int ix0c = clampi(ix0, 0, IMGW - 1), ix1c = clampi(ix1, 0, IMGW - 1);
  int iy0c = clampi(iy0, 0, IMGH - 1), iy1c = clampi(iy1, 0, IMGH - 1);
  int o00 = iy0c * IMGW + ix0c, o10 = iy0c * IMGW + ix1c;
  int o01 = iy1c * IMGW + ix0c, o11 = iy1c * IMGW + ix1c;

  const float* fb = img_features + (size_t)bn * DD * LPIX;
  _Float16* srow = S + (size_t)row * KAGG + (size_t)p * DD;
  for (int c = lane; c < DD; c += 32) {
    const float* fc = fb + (size_t)c * LPIX;
    float acc = fc[o00] * w00 + fc[o10] * w10 + fc[o01] * w01 + fc[o11] * w11;
    srow[c] = (_Float16)acc;
  }
}

// ---------------------------------------------------------------------------
// WMMA GEMM: C(M x N) = A(M x K, f16 row-major) * Bfrag(f16 fragment layout).
// Wave = 16 rows x (NACC*16) cols; block = 8 waves stacked along M (128 rows).
// B tiles double-buffered in LDS via async global->LDS copies (ASYNCcnt).
// All NACC B fragments are pulled into registers before the WMMA burst so the
// ds_loads pipeline and the WMMAs issue back-to-back.
// mode 0: outH = f16(relu(acc + bias))
// mode 1: feats += acc + bias ; outH = f16(acc + bias)
// mode 2: outF = acc + bias(col<realN)  (fp32)
// ---------------------------------------------------------------------------
template <int NACC>
__global__ __launch_bounds__(256) void gemm_kernel(
    const _Float16* __restrict__ A, const _Float16* __restrict__ Bf,
    const float* __restrict__ bias, int realN, int K, int nTilesB, int ldOut,
    int mode, _Float16* __restrict__ outH, float* __restrict__ feats,
    float* __restrict__ outF) {
  __shared__ __align__(32) _Float16 ldsB[2][NACC * 512];
  const int tid = threadIdx.x;
  const int wave = tid >> 5;
  const int lane = tid & 31;
  const int lrow = lane & 15;
  const int hi = lane >> 4;
  const int row0 = (blockIdx.y * 8 + wave) * 16;
  const int colBase = blockIdx.x * (NACC * 16);

  int rowA = row0 + lrow;
  if (rowA > MROWS - 1) rowA = MROWS - 1;
  const _Float16* arow = A + (size_t)rowA * K + hi * 16;

  const bool stager = (tid < NACC * 64);
  const _Float16* bBase = Bf + (size_t)blockIdx.x * NACC * 512 + (size_t)tid * 8;

  auto stage = [&](int kt, int buf) {
    const _Float16* src = bBase + (size_t)kt * nTilesB * 512;
    _Float16* dst = &ldsB[buf][tid * 8];
#if USE_ASYNC_LDS
    __builtin_amdgcn_global_load_async_to_lds_b128(
        (g_i32x4*)src, (l_i32x4*)dst, 0, 0);
#else
    *(u32x4*)dst = *(const u32x4*)src;
#endif
  };

  v8f acc[NACC] = {};
  const int ntk = K >> 5;

  if (stager) stage(0, 0);
  async_copy_fence();
  __syncthreads();

  for (int kt = 0; kt < ntk; ++kt) {
    const int buf = kt & 1;
    if (stager && (kt + 1 < ntk)) stage(kt + 1, buf ^ 1);  // overlap with compute

    union { v16h v; u32x4 q[2]; } af;
    const _Float16* ap = arow + kt * 32;
    af.q[0] = *(const u32x4*)(ap);
    af.q[1] = *(const u32x4*)(ap + 8);
    __builtin_prefetch(ap + 32, 0, 1);  // global_prefetch_b8 next K-tile

    v16h bfr[NACC];
#pragma unroll
    for (int j = 0; j < NACC; ++j)
      bfr[j] = *(const v16h*)(&ldsB[buf][j * 512 + lane * 16]);
#pragma unroll
    for (int j = 0; j < NACC; ++j)
      acc[j] = __builtin_amdgcn_wmma_f32_16x16x32_f16(
          false, af.v, false, bfr[j], (short)0, acc[j], false, false);

    async_copy_fence();   // next-buffer staging complete before barrier
    __syncthreads();      // everyone done reading buf before it is restaged
  }

  const int crow0 = row0 + hi * 8;
#pragma unroll
  for (int j = 0; j < NACC; ++j) {
    int ccol = colBase + j * 16 + lrow;
    float bv = (ccol < realN) ? bias[ccol] : 0.0f;
#pragma unroll
    for (int r = 0; r < 8; ++r) {
      int crow = crow0 + r;
      if (crow < MROWS) {
        float v = acc[j][r] + bv;
        if (mode == 0) {
          outH[(size_t)crow * ldOut + ccol] = (_Float16)fmaxf(v, 0.0f);
        } else if (mode == 1) {
          feats[(size_t)crow * DD + ccol] += v;
          outH[(size_t)crow * ldOut + ccol] = (_Float16)v;
        } else {
          outF[(size_t)crow * ldOut + ccol] = v;
        }
      }
    }
  }
}

// ---------------------------------------------------------------------------
// Mean update (logit-space sigmoid) + covariance blend.
// ---------------------------------------------------------------------------
__global__ void update_kernel(const float* __restrict__ delta,
                              const float* __restrict__ wlog,
                              const float* __restrict__ dirv,
                              float* __restrict__ meanbuf, float* __restrict__ covbuf) {
  int row = blockIdx.x * blockDim.x + threadIdx.x;
  if (row >= MROWS) return;
  const float lo[3]   = {-50.f, -50.f, -10.f};
  const float span[3] = {100.f, 100.f, 20.f};
#pragma unroll
  for (int i = 0; i < 3; ++i) {
    float m = meanbuf[(size_t)row * 3 + i];
    float mn = fminf(fmaxf((m - lo[i]) / span[i], 0.f), 1.f);
    float a = fmaxf(mn, 1e-5f);
    float bq = fmaxf(1.f - mn, 1e-5f);
    float z = __logf(a / bq) + delta[(size_t)row * 16 + i];
    float s = 1.f / (1.f + __expf(-z));
    meanbuf[(size_t)row * 3 + i] = s * span[i] + lo[i];
  }
  float w = 1.f / (1.f + __expf(-wlog[(size_t)row * 16]));
  float dx = dirv[(size_t)row * 3 + 0], dy = dirv[(size_t)row * 3 + 1];
  float vx = dy, vy = -dx;  // cross(dirv, e_z)
  float nrm = fmaxf(sqrtf(vx * vx + vy * vy), 1e-12f);
  vx /= nrm; vy /= nrm;
  float vv[3] = {vx, vy, 0.f};
#pragma unroll
  for (int i = 0; i < 3; ++i)
#pragma unroll
    for (int j = 0; j < 3; ++j) {
      size_t o = (size_t)row * 9 + i * 3 + j;
      covbuf[o] = covbuf[o] * w + vv[i] * vv[j] * (1.f - w);
    }
}

__global__ void writeout_kernel(const float* __restrict__ feats,
                                const float* __restrict__ meanbuf,
                                const float* __restrict__ covbuf,
                                float* __restrict__ out) {
  size_t total = (size_t)FEATS_N + MEAN_N + COV_N;
  for (size_t i = (size_t)blockIdx.x * blockDim.x + threadIdx.x; i < total;
       i += (size_t)gridDim.x * blockDim.x) {
    float v;
    if (i < FEATS_N) v = feats[i];
    else if (i < FEATS_N + MEAN_N) v = meanbuf[i - FEATS_N];
    else v = covbuf[i - FEATS_N - MEAN_N];
    out[i] = v;
  }
}

// ---------------------------------------------------------------------------
extern "C" void kernel_launch(void* const* d_in, const int* in_sizes, int n_in,
                              void* d_out, int out_size, void* d_ws, size_t ws_size,
                              hipStream_t stream) {
  (void)in_sizes; (void)n_in; (void)out_size; (void)ws_size;
  const float* img_features = (const float*)d_in[0];
  const float* means3D      = (const float*)d_in[1];
  const float* features     = (const float*)d_in[2];
  const float* cov3D        = (const float*)d_in[3];
  const float* dirv         = (const float*)d_in[4];
  const float* lidar2img    = (const float*)d_in[6];
  const float* agg_w1  = (const float*)d_in[7];
  const float* agg_b1  = (const float*)d_in[8];
  const float* agg_w2  = (const float*)d_in[9];
  const float* agg_b2  = (const float*)d_in[10];
  const float* mean_w1 = (const float*)d_in[11];
  const float* mean_b1 = (const float*)d_in[12];
  const float* mean_w2 = (const float*)d_in[13];
  const float* mean_b2 = (const float*)d_in[14];
  const float* var_w1  = (const float*)d_in[15];
  const float* var_b1  = (const float*)d_in[16];
  const float* var_w2  = (const float*)d_in[17];
  const float* var_b2  = (const float*)d_in[18];
  float* out = (float*)d_out;

  char* ws = (char*)d_ws;
  size_t off = 0;
  auto alloc = [&](size_t bytes) -> char* {
    char* p = ws + off;
    off = (off + bytes + 255) & ~(size_t)255;
    return p;
  };
  _Float16* W1h  = (_Float16*)alloc((size_t)KAGG * DD * 2);
  _Float16* W2h  = (_Float16*)alloc((size_t)DD * DD * 2);
  _Float16* MW1h = (_Float16*)alloc((size_t)DD * DD * 2);
  _Float16* MW2h = (_Float16*)alloc((size_t)DD * 16 * 2);
  _Float16* VW1h = (_Float16*)alloc((size_t)DD * DD * 2);
  _Float16* VW2h = (_Float16*)alloc((size_t)DD * 16 * 2);
  _Float16* S    = (_Float16*)alloc((size_t)MROWS * KAGG * 2);
  _Float16* Y1h  = (_Float16*)alloc((size_t)MROWS * DD * 2);
  _Float16* Smh  = (_Float16*)alloc((size_t)MROWS * DD * 2);
  float* featsbuf = (float*)alloc((size_t)MROWS * DD * 4);
  float* meanbuf  = (float*)alloc((size_t)MROWS * 3 * 4);
  float* covbuf   = (float*)alloc((size_t)MROWS * 9 * 4);
  float* deltab   = (float*)alloc((size_t)MROWS * 16 * 4);
  float* wlogb    = (float*)alloc((size_t)MROWS * 16 * 4);

  {
    int tot = KAGG * DD;
    convert_wfrag<<<(tot + 255) / 256, 256, 0, stream>>>(agg_w1, W1h, KAGG, DD, DD);
    tot = DD * DD;
    convert_wfrag<<<(tot + 255) / 256, 256, 0, stream>>>(agg_w2, W2h, DD, DD, DD);
    convert_wfrag<<<(tot + 255) / 256, 256, 0, stream>>>(mean_w1, MW1h, DD, DD, DD);
    convert_wfrag<<<(tot + 255) / 256, 256, 0, stream>>>(var_w1, VW1h, DD, DD, DD);
    tot = DD * 16;
    convert_wfrag<<<(tot + 255) / 256, 256, 0, stream>>>(mean_w2, MW2h, DD, 3, 16);
    convert_wfrag<<<(tot + 255) / 256, 256, 0, stream>>>(var_w2, VW2h, DD, 1, 16);
  }
  init_kernel<<<MROWS, 256, 0, stream>>>(features, means3D, cov3D, featsbuf, meanbuf, covbuf);

  dim3 gBig(4, (MROWS / 16 + 7) / 8);    // N=256: 4 col-blocks of 64 cols
  dim3 gSmall(1, (MROWS / 16 + 7) / 8);  // N=16 (padded heads)
  float* pts_base = out + (size_t)FEATS_N + MEAN_N + COV_N;

  for (int layer = 0; layer < 4; ++layer) {
    gather_kernel<<<MROWS, 256, 0, stream>>>(img_features, meanbuf, covbuf,
                                             lidar2img, S, pts_base, layer);
    gemm_kernel<4><<<gBig, 256, 0, stream>>>(S, W1h, agg_b1, 256, KAGG, 16, 256, 0,
                                             Y1h, nullptr, nullptr);
    gemm_kernel<4><<<gBig, 256, 0, stream>>>(Y1h, W2h, agg_b2, 256, 256, 16, 256, 1,
                                             Smh, featsbuf, nullptr);
    gemm_kernel<4><<<gBig, 256, 0, stream>>>(Smh, MW1h, mean_b1, 256, 256, 16, 256, 0,
                                             Y1h, nullptr, nullptr);
    gemm_kernel<1><<<gSmall, 256, 0, stream>>>(Y1h, MW2h, mean_b2, 3, 256, 1, 16, 2,
                                               nullptr, nullptr, deltab);
    gemm_kernel<4><<<gBig, 256, 0, stream>>>(Smh, VW1h, var_b1, 256, 256, 16, 256, 0,
                                             Y1h, nullptr, nullptr);
    gemm_kernel<1><<<gSmall, 256, 0, stream>>>(Y1h, VW2h, var_b2, 1, 256, 1, 16, 2,
                                               nullptr, nullptr, wlogb);
    update_kernel<<<(MROWS + 255) / 256, 256, 0, stream>>>(deltab, wlogb, dirv,
                                                           meanbuf, covbuf);
  }
  writeout_kernel<<<2048, 256, 0, stream>>>(featsbuf, meanbuf, covbuf, out);
}
